// CausalSelfAttention_35089882808734
// MI455X (gfx1250) — matrix operand order repaired and checked by reading, not compiled, not run on hardware
//
#include <hip/hip_runtime.h>

// ---------------------------------------------------------------------------
// Problem constants (reference): B=1, T=4096, C=1024, H=16, HKV=4, D=64
// ---------------------------------------------------------------------------
#define T_LEN   4096
#define C_DIM   1024
#define NH      16
#define NKV     4
#define HD      64
#define GATECH  32

typedef __bf16 bf16;
typedef __attribute__((ext_vector_type(16))) __bf16    v16bf;
typedef __attribute__((ext_vector_type(8)))  float     v8f;
typedef __attribute__((ext_vector_type(4)))  unsigned  u32x4;
typedef __attribute__((ext_vector_type(8)))  unsigned  u32x8;

union BF16x16 {
  v16bf v;
  uint4 q[2];
};

__device__ __forceinline__ bf16 f2bf(float f) {
  union { float f; unsigned u; } a; a.f = f;
  unsigned r = a.u + 0x7FFFu + ((a.u >> 16) & 1u);   // round-to-nearest-even
  union { unsigned short s; bf16 b; } o;
  o.s = (unsigned short)(r >> 16);
  return o.b;
}

__device__ __forceinline__ v8f wmma_bf16(v16bf a, v16bf b, v8f c) {
  return __builtin_amdgcn_wmma_f32_16x16x32_bf16(false, a, false, b, (short)0, c,
                                                 false, false);
}

// ---------------------------------------------------------------------------
// Tensor Data Mover: 2D tile (tile_w x tile_h, bf16) global -> LDS.
// D# per CDNA5 ISA (group0 128b + group1 256b; groups 2/3
// omitted => 2D tensor). Wave-uniform issue, EXEC ignored, TENSORcnt tracked.
// ---------------------------------------------------------------------------
__device__ __forceinline__ void tdm_load_2d_bf16(unsigned lds_off, const void* gptr,
                                                 unsigned tile_w, unsigned tile_h,
                                                 unsigned tensor_w, unsigned tensor_h,
                                                 unsigned stride_elems) {
  unsigned long long ga = (unsigned long long)gptr;
  u32x4 g0;
  g0[0] = 1u;                                           // count=1 valid user D#
  g0[1] = lds_off;                                      // lds_addr (bytes)
  g0[2] = (unsigned)(ga & 0xffffffffull);               // global_addr[31:0]
  g0[3] = (unsigned)((ga >> 32) & 0x01ffffffull)        // global_addr[56:32]
          | 0x80000000u;                                // type = 2 ("image")
  u32x8 g1;
  g1[0] = (1u << 16);                                   // data_size=1 -> 2 bytes
  g1[1] = (tensor_w & 0xffffu) << 16;                   // tensor_dim0[15:0]
  g1[2] = ((tensor_w >> 16) & 0xffffu)                  // tensor_dim0[31:16]
          | ((tensor_h & 0xffffu) << 16);               // tensor_dim1[15:0]
  g1[3] = ((tensor_h >> 16) & 0xffffu)                  // tensor_dim1[31:16]
          | (tile_w << 16);                             // tile_dim0
  g1[4] = tile_h;                                       // tile_dim1 (tile_dim2=0)
  g1[5] = stride_elems;                                 // tensor_dim0_stride[31:0]
  g1[6] = 0u;
  g1[7] = 0u;
  asm volatile("tensor_load_to_lds %0, %1" :: "s"(g0), "s"(g1) : "memory");
}

// Per-lane async copy global -> LDS (128b), tracked by ASYNCcnt.
__device__ __forceinline__ void async_b128(unsigned lds_off, const void* g) {
  asm volatile("global_load_async_to_lds_b128 %0, %1, off"
               :: "v"(lds_off), "v"(g) : "memory");
}
__device__ __forceinline__ void wait_async0() {
  asm volatile("s_wait_asynccnt 0x0" ::: "memory");
}

// ---------------------------------------------------------------------------
// f32 -> bf16 elementwise
// ---------------------------------------------------------------------------
__global__ void k_cvt(const float* __restrict__ src, bf16* __restrict__ dst, int n) {
  int i = blockIdx.x * blockDim.x + threadIdx.x;
  int stride = gridDim.x * blockDim.x;
  for (; i < n; i += stride) dst[i] = f2bf(src[i]);
}

// f32 (1024 x N, row-major) -> bf16 transposed (N x 1024, row-major)
__global__ void k_cvtT(const float* __restrict__ src, bf16* __restrict__ dst, int N) {
  int i = blockIdx.x * blockDim.x + threadIdx.x;
  int stride = gridDim.x * blockDim.x;
  int total = N * C_DIM;
  for (; i < total; i += stride) {
    int n = i >> 10, k = i & (C_DIM - 1);       // K == C_DIM == 1024
    dst[i] = f2bf(src[(size_t)k * N + n]);
  }
}

// ---------------------------------------------------------------------------
// GEMM: C(f32, MxN) = A(bf16, MxK row-major) @ Bt(bf16, NxK row-major)^T
// Block tile 128x128, 8 waves, wave tile 32x64 (8 WMMAs/K-step).
// Double-buffered LDS tiles filled by the Tensor Data Mover so the DMA of
// tile t+1 overlaps the WMMAs of tile t.
// ---------------------------------------------------------------------------
__global__ __launch_bounds__(256) void k_gemm(const bf16* __restrict__ A,
                                              const bf16* __restrict__ Bt,
                                              float* __restrict__ Cm,
                                              int M, int N, int K) {
  __shared__ __align__(16) bf16 sA[2][128 * 32];    // [buf][m][k]
  __shared__ __align__(16) bf16 sBT[2][128 * 32];   // [buf][n][k]

  const int tid  = threadIdx.x;
  const int lane = tid & 31;
  const int wave = tid >> 5;
  const int hl   = lane >> 4;
  const int r16  = lane & 15;
  const int m_off = (wave >> 1) * 32;   // 0,32,64,96
  const int n_off = (wave & 1) * 64;    // 0,64
  const int bm = blockIdx.y * 128;
  const int bn = blockIdx.x * 128;

  v8f acc[2][4] = {};

  // prologue: DMA tile 0 into buffer 0
  if (wave == 0) {
    tdm_load_2d_bf16((unsigned)(size_t)&sA[0][0], A + (size_t)bm * K,
                     32, 128, (unsigned)K, (unsigned)M, (unsigned)K);
    tdm_load_2d_bf16((unsigned)(size_t)&sBT[0][0], Bt + (size_t)bn * K,
                     32, 128, (unsigned)K, (unsigned)N, (unsigned)K);
    __builtin_amdgcn_s_wait_tensorcnt(0);
  }
  __syncthreads();

  int buf = 0;
  for (int k0 = 0; k0 < K; k0 += 32) {
    const bool has_next = (k0 + 32) < K;
    if (wave == 0 && has_next) {          // DMA next tile into other buffer
      tdm_load_2d_bf16((unsigned)(size_t)&sA[buf ^ 1][0],
                       A + (size_t)bm * K + k0 + 32,
                       32, 128, (unsigned)K, (unsigned)M, (unsigned)K);
      tdm_load_2d_bf16((unsigned)(size_t)&sBT[buf ^ 1][0],
                       Bt + (size_t)bn * K + k0 + 32,
                       32, 128, (unsigned)K, (unsigned)N, (unsigned)K);
    }

    const bf16* cA = sA[buf];
    const bf16* cB = sBT[buf];
    BF16x16 af[2], bfg[4];
#pragma unroll
    for (int mi = 0; mi < 2; ++mi) {
      const bf16* p = cA + (m_off + mi * 16 + r16) * 32 + hl * 8;
      af[mi].q[0] = *(const uint4*)(p);        // k = base .. base+7
      af[mi].q[1] = *(const uint4*)(p + 16);   // k = base+16 .. base+23
    }
#pragma unroll
    for (int ni = 0; ni < 4; ++ni) {
      const bf16* p = cB + (n_off + ni * 16 + r16) * 32 + hl * 8;  // k=8*hl+i, 16+8*hl+i
      bfg[ni].q[0] = *(const uint4*)(p);
      bfg[ni].q[1] = *(const uint4*)(p + 16);
    }
#pragma unroll
    for (int mi = 0; mi < 2; ++mi)
#pragma unroll
      for (int ni = 0; ni < 4; ++ni)
        acc[mi][ni] = wmma_bf16(af[mi].v, bfg[ni].v, acc[mi][ni]);

    if (wave == 0 && has_next) __builtin_amdgcn_s_wait_tensorcnt(0);
    __syncthreads();
    buf ^= 1;
  }

  // epilogue: base pointer + constant-offset stores (no per-store 64b muls)
  float* cbase = Cm + (size_t)(bm + m_off + 8 * hl) * N + bn + n_off + r16;
#pragma unroll
  for (int mi = 0; mi < 2; ++mi)
#pragma unroll
    for (int r = 0; r < 8; ++r) {
      float* cr = cbase + (size_t)(mi * 16 + r) * N;
#pragma unroll
      for (int ni = 0; ni < 4; ++ni) cr[ni * 16] = acc[mi][ni][r];
    }
}

// ---------------------------------------------------------------------------
// v = vraw + 2*sigmoid(x[:, :32] @ Wg) * ve  ->  bf16, channel-major layout
// vT[(kvh*64 + d) * T + t]. One thread per (t, kv-head).
// ---------------------------------------------------------------------------
__global__ void k_gate_v(const float* __restrict__ x, const float* __restrict__ ve,
                         const float* __restrict__ vraw, const float* __restrict__ Wg,
                         bf16* __restrict__ vT) {
  int idx = blockIdx.x * blockDim.x + threadIdx.x;
  if (idx >= T_LEN * NKV) return;
  int t  = idx >> 2;
  int hk = idx & 3;
  const float* xr = x + (size_t)t * C_DIM;
  float acc = 0.f;
#pragma unroll
  for (int c = 0; c < GATECH; ++c) acc += xr[c] * Wg[c * NKV + hk];
  float gate = 2.f / (1.f + __expf(-acc));
  size_t base = (size_t)t * (NKV * HD) + hk * HD;
#pragma unroll
  for (int d = 0; d < HD; ++d)
    vT[(size_t)(hk * HD + d) * T_LEN + t] = f2bf(vraw[base + d] + gate * ve[base + d]);
}

// ---------------------------------------------------------------------------
// RoPE then RMS-norm over D=64, f32 -> bf16. One thread per (t, head).
// ---------------------------------------------------------------------------
template <int NHh>
__global__ void k_rope_norm(const float* __restrict__ src, bf16* __restrict__ dst,
                            const float* __restrict__ cosb,
                            const float* __restrict__ sinb) {
  int idx = blockIdx.x * blockDim.x + threadIdx.x;
  if (idx >= T_LEN * NHh) return;
  int t = idx / NHh, hh = idx % NHh;      // NHh is a power of two -> shifts
  const float* xr = src + ((size_t)t * NHh + hh) * HD;
  const float* cs = cosb + (size_t)t * (HD / 2);
  const float* sn = sinb + (size_t)t * (HD / 2);
  float o[HD];
#pragma unroll
  for (int j = 0; j < HD / 2; ++j) {
    float x1 = xr[j], x2 = xr[j + HD / 2];
    o[j]          =  x1 * cs[j] + x2 * sn[j];
    o[j + HD / 2] = -x1 * sn[j] + x2 * cs[j];
  }
  float ms = 0.f;
#pragma unroll
  for (int j = 0; j < HD; ++j) ms += o[j] * o[j];
  float rs = rsqrtf(ms * (1.f / HD) + 1.1920929e-7f);
  bf16* dr = dst + ((size_t)t * NHh + hh) * HD;
#pragma unroll
  for (int j = 0; j < HD; ++j) dr[j] = f2bf(o[j] * rs);
}

// ---------------------------------------------------------------------------
// Flash attention, causal + sliding window. 4 waves/block, each wave owns one
// 16-row q tile of one head; block shares double-buffered async K/V staging.
// grid = (T/64, H), block = 128.
// ---------------------------------------------------------------------------
__global__ __launch_bounds__(128) void k_attn(const bf16* __restrict__ Qn,
                                              const bf16* __restrict__ Kn,
                                              const bf16* __restrict__ vT,
                                              bf16* __restrict__ Y,
                                              const int* __restrict__ wsz_p) {
  __shared__ __align__(16) bf16 sK[2][32 * 64];   // [buf][key][ch]
  __shared__ __align__(16) bf16 sVT[2][64 * 32];  // [buf][ch][key]
  __shared__ __align__(16) bf16 sP[4][16 * 32];   // per-wave P staging

  const int tid  = threadIdx.x;
  const int lane = tid & 31;
  const int wave = tid >> 5;
  const int hl   = lane >> 4;
  const int r16  = lane & 15;
  const int h    = blockIdx.y;
  const int kvh  = h >> 2;                        // GQA group of 4
  const int rowB = blockIdx.x * 64;               // block's first q row
  const int row0 = rowB + wave * 16;              // this wave's q tile
  const int wsz  = *wsz_p;
  const bool use_win = (wsz > 0) && (wsz < T_LEN);
  const float scale = 0.125f;                     // 1/sqrt(64)

  auto stage = [&](int kb, bf16* dK, bf16* dV) {
#pragma unroll
    for (int u = 0; u < 2; ++u) {                 // K: 256 16B-chunks
      const int c   = tid * 2 + u;
      const int krw = c >> 3;                     // key row 0..31
      const int co  = (c & 7) * 8;                // channel offset
      async_b128((unsigned)(size_t)(dK + krw * 64 + co),
                 Kn + (size_t)(kb + krw) * (NKV * HD) + kvh * HD + co);
    }
#pragma unroll
    for (int u = 0; u < 2; ++u) {                 // V^T: 256 16B-chunks
      const int c  = tid * 2 + u;
      const int ch = c >> 2;                      // channel 0..63
      const int ko = (c & 3) * 8;                 // key offset
      async_b128((unsigned)(size_t)(dV + ch * 32 + ko),
                 vT + (size_t)(kvh * HD + ch) * T_LEN + kb + ko);
    }
  };

  // Q A-fragments (K-dim 64 = two 32-chunks), k = 32*c + {8*hl+i, 16+8*hl+i}, from global
  BF16x16 qf[2];
  {
    const bf16* qrow = Qn + (size_t)(row0 + r16) * C_DIM + h * HD;
#pragma unroll
    for (int c = 0; c < 2; ++c) {
      const int base = c * 32 + hl * 8;
      qf[c].q[0] = *(const uint4*)(qrow + base);
      qf[c].q[1] = *(const uint4*)(qrow + base + 16);
    }
  }

  float m[8], l[8];
  v8f acc[4];
#pragma unroll
  for (int r = 0; r < 8; ++r) { m[r] = -INFINITY; l[r] = 0.f; }
  {
    v8f z = {0.f, 0.f, 0.f, 0.f, 0.f, 0.f, 0.f, 0.f};
#pragma unroll
    for (int ni = 0; ni < 4; ++ni) acc[ni] = z;
  }

  int lo = use_win ? (rowB - wsz) : 0;
  if (lo < 0) lo = 0;
  const int kb0  = lo & ~31;
  const int kend = rowB + 64;

  // prologue: stage first tile into buffer 0
  stage(kb0, sK[0], sVT[0]);
  wait_async0();
  __syncthreads();

  int buf = 0;
  for (int kb = kb0; kb < kend; kb += 32) {
    if (kb + 32 < kend) stage(kb + 32, sK[buf ^ 1], sVT[buf ^ 1]);

    const bool active =
        (kb <= row0 + 15) && (!use_win || (kb + 31 >= row0 - wsz));
    if (active) {
      const bf16* cK = sK[buf];
      const bf16* cV = sVT[buf];
      // ---- S = Q @ K^T : two 16x16 score blocks
      v8f s[2];
#pragma unroll
      for (int tc = 0; tc < 2; ++tc) {
        v8f sv = {0.f, 0.f, 0.f, 0.f, 0.f, 0.f, 0.f, 0.f};
#pragma unroll
        for (int c = 0; c < 2; ++c) {
          BF16x16 kt;   // B-frag of K^T: col=key(r16), k=ch = 32*c + {8*hl+i, 16+8*hl+i}
          const bf16* krow = cK + (tc * 16 + r16) * 64 + c * 32 + hl * 8;
          kt.q[0] = *(const uint4*)(krow);
          kt.q[1] = *(const uint4*)(krow + 16);
          sv = wmma_bf16(qf[c].v, kt.v, sv);
        }
        s[tc] = sv;
      }

      // ---- mask + scale + online softmax update
#pragma unroll
      for (int r = 0; r < 8; ++r) {
        const int row = row0 + r + 8 * hl;
        const int k0c = kb + r16;
        const int k1c = kb + 16 + r16;
        const bool ok0 = (k0c <= row) && (!use_win || k0c >= row - wsz);
        const bool ok1 = (k1c <= row) && (!use_win || k1c >= row - wsz);
        float v0 = ok0 ? s[0][r] * scale : -INFINITY;
        float v1 = ok1 ? s[1][r] * scale : -INFINITY;

        float mx = fmaxf(v0, v1);
#pragma unroll
        for (int o = 8; o >= 1; o >>= 1) mx = fmaxf(mx, __shfl_xor(mx, o, 32));
        const float mnew = fmaxf(m[r], mx);

        const float alpha = (mnew == -INFINITY) ? 1.f : __expf(m[r] - mnew);
        const float p0 = (v0 == -INFINITY) ? 0.f : __expf(v0 - mnew);
        const float p1 = (v1 == -INFINITY) ? 0.f : __expf(v1 - mnew);
        float rsum = p0 + p1;
#pragma unroll
        for (int o = 8; o >= 1; o >>= 1) rsum += __shfl_xor(rsum, o, 32);
        l[r] = l[r] * alpha + rsum;
        m[r] = mnew;
#pragma unroll
        for (int ni = 0; ni < 4; ++ni) acc[ni][r] = acc[ni][r] * alpha;

        sP[wave][(r + 8 * hl) * 32 + r16]      = f2bf(p0);
        sP[wave][(r + 8 * hl) * 32 + 16 + r16] = f2bf(p1);
      }

      // ---- O += P(16x32) @ V(32x64)
      BF16x16 pa;
      {
        const bf16* prow = &sP[wave][r16 * 32 + hl * 8];
        pa.q[0] = *(const uint4*)(prow);        // keys base .. base+7
        pa.q[1] = *(const uint4*)(prow + 16);   // keys base+16 .. base+23
      }
#pragma unroll
      for (int ni = 0; ni < 4; ++ni) {
        BF16x16 vf;   // B-frag of V: col=ch(ni*16+r16), k=key = {8*hl+i, 16+8*hl+i}
        const bf16* vrow = cV + (ni * 16 + r16) * 32 + hl * 8;
        vf.q[0] = *(const uint4*)(vrow);
        vf.q[1] = *(const uint4*)(vrow + 16);
        acc[ni] = wmma_bf16(pa.v, vf.v, acc[ni]);
      }
    }

    wait_async0();       // next buffer fully in LDS
    __syncthreads();
    buf ^= 1;
  }

  // ---- epilogue: normalize, write bf16 y
  float inv[8];
#pragma unroll
  for (int r = 0; r < 8; ++r) inv[r] = (l[r] > 0.f) ? (1.f / l[r]) : 0.f;
  bf16* ybase = Y + (size_t)(row0 + 8 * hl) * C_DIM + h * HD + r16;
#pragma unroll
  for (int r = 0; r < 8; ++r) {
    bf16* yr = ybase + (size_t)r * C_DIM;
#pragma unroll
    for (int ni = 0; ni < 4; ++ni) yr[ni * 16] = f2bf(acc[ni][r] * inv[r]);
  }
}

// ---------------------------------------------------------------------------
// Host-side orchestration
// ---------------------------------------------------------------------------
extern "C" void kernel_launch(void* const* d_in, const int* in_sizes, int n_in,
                              void* d_out, int out_size, void* d_ws, size_t ws_size,
                              hipStream_t stream) {
  const float* x    = (const float*)d_in[0];
  const float* ve   = (const float*)d_in[1];
  const float* cosb = (const float*)d_in[2];
  const float* sinb = (const float*)d_in[3];
  const float* Wq   = (const float*)d_in[4];
  const float* Wk   = (const float*)d_in[5];
  const float* Wv   = (const float*)d_in[6];
  const float* Wo   = (const float*)d_in[7];
  const float* Wg   = (const float*)d_in[8];
  const int*   wsz  = (const int*)d_in[9];

  char* ws = (char*)d_ws;
  size_t off = 0;
  auto carve = [&](size_t bytes) -> char* {
    char* p = ws + off;
    off += (bytes + 255) & ~(size_t)255;
    return p;
  };

  const size_t nX  = (size_t)T_LEN * C_DIM;
  const size_t nQ  = (size_t)T_LEN * NH * HD;
  const size_t nKV = (size_t)T_LEN * NKV * HD;
  const size_t nWq = (size_t)C_DIM * NH * HD;
  const size_t nWk = (size_t)C_DIM * NKV * HD;
  const size_t nWo = (size_t)C_DIM * C_DIM;

  bf16*  xb   = (bf16*)carve(nX  * 2);
  bf16*  wqt  = (bf16*)carve(nWq * 2);   // transposed (N x K)
  bf16*  wkt  = (bf16*)carve(nWk * 2);
  bf16*  wvt  = (bf16*)carve(nWk * 2);
  bf16*  wot  = (bf16*)carve(nWo * 2);
  float* qraw = (float*)carve(nQ  * 4);
  float* kraw = (float*)carve(nKV * 4);
  float* vraw = (float*)carve(nKV * 4);
  bf16*  qn   = (bf16*)carve(nQ  * 2);
  bf16*  kn   = (bf16*)carve(nKV * 2);
  bf16*  vt   = (bf16*)carve(nKV * 2);   // channel-major V
  bf16*  yb   = (bf16*)carve(nX  * 2);

  // 1) conversions: x plain, weights transposed to (N x K) bf16
  k_cvt <<<dim3(2048), dim3(256), 0, stream>>>(x,  xb,  (int)nX);
  k_cvtT<<<dim3(2048), dim3(256), 0, stream>>>(Wq, wqt, NH * HD);
  k_cvtT<<<dim3(512),  dim3(256), 0, stream>>>(Wk, wkt, NKV * HD);
  k_cvtT<<<dim3(512),  dim3(256), 0, stream>>>(Wv, wvt, NKV * HD);
  k_cvtT<<<dim3(2048), dim3(256), 0, stream>>>(Wo, wot, C_DIM);

  // 2) Q/K/V projections (double-buffered TDM WMMA GEMM)
  k_gemm<<<dim3((NH * HD) / 128, T_LEN / 128), dim3(256), 0, stream>>>(
      xb, wqt, qraw, T_LEN, NH * HD, C_DIM);
  k_gemm<<<dim3((NKV * HD) / 128, T_LEN / 128), dim3(256), 0, stream>>>(
      xb, wkt, kraw, T_LEN, NKV * HD, C_DIM);
  k_gemm<<<dim3((NKV * HD) / 128, T_LEN / 128), dim3(256), 0, stream>>>(
      xb, wvt, vraw, T_LEN, NKV * HD, C_DIM);

  // 3) gated value-embedding add -> channel-major bf16 V
  k_gate_v<<<dim3((T_LEN * NKV + 255) / 256), dim3(256), 0, stream>>>(
      x, ve, vraw, Wg, vt);

  // 4) RoPE + RMS-norm on q and k
  k_rope_norm<NH> <<<dim3((T_LEN * NH + 255) / 256), dim3(256), 0, stream>>>(
      qraw, qn, cosb, sinb);
  k_rope_norm<NKV><<<dim3((T_LEN * NKV + 255) / 256), dim3(256), 0, stream>>>(
      kraw, kn, cosb, sinb);

  // 5) sliding-window causal flash attention (WMMA + double-buffered async)
  k_attn<<<dim3(T_LEN / 64, NH), dim3(128), 0, stream>>>(qn, kn, vt, yb, wsz);

  // 6) output projection -> f32 d_out
  k_gemm<<<dim3(C_DIM / 128, T_LEN / 128), dim3(256), 0, stream>>>(
      yb, wot, (float*)d_out, T_LEN, C_DIM, C_DIM);
}
